// Decoder_39324720562636
// MI455X (gfx1250) — matrix operand order repaired and checked by reading, not compiled, not run on hardware
//
#include <hip/hip_runtime.h>
#include <hip/hip_bf16.h>

// ---------------------------------------------------------------------------
// Problem constants (from reference): B=4, C=256, HEADS=8, H2=W2=128
// ---------------------------------------------------------------------------
#define NB     4
#define NC     256
#define NHEAD  8
#define HC     32          // channels per head
#define H2     128
#define W2     128
#define NPIX   16384       // H2*W2
#define LOG2PIX 14
#define HIN    64
#define WIN    64

typedef __bf16 bf16_t;
typedef __attribute__((ext_vector_type(16))) __bf16 v16bf;
typedef __attribute__((ext_vector_type(8)))  __bf16 v8bf;
typedef __attribute__((ext_vector_type(8)))  float  v8f;

// ---------------------------------------------------------------------------
// WMMA helpers (CDNA5 wave32, 16x16x32 bf16 -> f32)
// Fragment layouts per CDNA5 ISA 7.12.2:
//  A (16x32 MxK): lane m=L%16; L<16 -> elems0..7 = K k0..k0+7, elems8..15 = K k0+16..k0+23
//                             L>=16 -> elems0..7 = K k0+8..k0+15, elems8..15 = K k0+24..k0+31
//  B (32x16 KxN): lane n=L%16; same K layout as A (L<16: k0..k0+7, k0+16..k0+23; L>=16: k0+8..k0+15, k0+24..k0+31)
//  C/D (16x16 f32): VGPR r: lanes0-15 -> (M=r, N=lane); lanes16-31 -> (M=8+r, N=lane-16)
// ---------------------------------------------------------------------------
__device__ __forceinline__ v16bf load_a_frag(const bf16_t* __restrict__ A, int lda,
                                             int m0, int k0, int lane) {
    int m  = m0 + (lane & 15);
    int ko = (lane < 16) ? 0 : 8;
    const bf16_t* p = A + (size_t)m * lda + k0 + ko;
    v16bf f;
    *(v8bf*)&f        = *(const v8bf*)p;          // K = k0+ko .. +7
    *(((v8bf*)&f) + 1) = *(const v8bf*)(p + 16);  // K = k0+16+ko .. +7
    return f;
}

__device__ __forceinline__ v16bf load_b_frag(const bf16_t* __restrict__ BT, int ldb,
                                             int n0, int k0, int lane) {
    int n  = n0 + (lane & 15);
    int ko = (lane < 16) ? 0 : 8;
    const bf16_t* p = BT + (size_t)n * ldb + k0 + ko;
    v16bf f;
    *(v8bf*)&f        = *(const v8bf*)p;          // K = k0+ko .. +7
    *(((v8bf*)&f) + 1) = *(const v8bf*)(p + 16);  // K = k0+16+ko .. +7
    return f;
}

__device__ __forceinline__ v8f wmma_bf16(v16bf a, v16bf b, v8f c) {
    return __builtin_amdgcn_wmma_f32_16x16x32_bf16(
        /*neg_a=*/false, a, /*neg_b=*/false, b,
        /*c_mod=*/(short)0, c, /*reuse_a=*/false, /*reuse_b=*/false);
}

// ---------------------------------------------------------------------------
// Bilinear x2 upsample tap (align_corners=False / half-pixel, edge clamp)
// out coord i -> src = i*0.5 - 0.25; even i: (k-1:0.25, k:0.75); odd: (k:0.75, k+1:0.25)
// ---------------------------------------------------------------------------
__device__ __forceinline__ float bilin_up(const float* __restrict__ xin,
                                          int b, int c, int h2, int w2) {
    int hl0 = (h2 - 1) >> 1;
    int wl0 = (w2 - 1) >> 1;
    float whl = (h2 & 1) ? 0.75f : 0.25f;
    float wwl = (w2 & 1) ? 0.75f : 0.25f;
    int hl = hl0 < 0 ? 0 : hl0;
    int hh = (hl0 + 1) > (HIN - 1) ? (HIN - 1) : (hl0 + 1);
    int wl = wl0 < 0 ? 0 : wl0;
    int wh = (wl0 + 1) > (WIN - 1) ? (WIN - 1) : (wl0 + 1);
    const float* p = xin + (((size_t)b * NC + c) << 12);   // 64*64 = 4096
    float v00 = p[(hl << 6) + wl], v01 = p[(hl << 6) + wh];
    float v10 = p[(hh << 6) + wl], v11 = p[(hh << 6) + wh];
    float whh = 1.0f - whl, wwh = 1.0f - wwl;
    return whl * (wwl * v00 + wwh * v01) + whh * (wwl * v10 + wwh * v11);
}

// ---------------------------------------------------------------------------
// prep: convert 5 weight matrices (256x256 f32) -> bf16; zero the gram buffer
// ---------------------------------------------------------------------------
__global__ void prep_kernel(const float* __restrict__ w0, const float* __restrict__ w1,
                            const float* __restrict__ w2, const float* __restrict__ w3,
                            const float* __restrict__ w4,
                            bf16_t* __restrict__ wbf, float* __restrict__ gram) {
    int idx = blockIdx.x * 256 + threadIdx.x;
    if (idx < 5 * 65536) {
        const float* s;
        int wi = idx >> 16, off = idx & 65535;
        switch (wi) {
            case 0: s = w0; break; case 1: s = w1; break;
            case 2: s = w2; break; case 3: s = w3; break;
            default: s = w4; break;
        }
        wbf[idx] = (bf16_t)s[off];
    } else {
        int g = idx - 5 * 65536;
        if (g < NB * NHEAD * HC * HC) gram[g] = 0.0f;
    }
}

// ---------------------------------------------------------------------------
// LayerNorm over channels per pixel + emit transposed bf16 copies:
//   rawT[b][n][c] (plain cast)  and  lnT[b][n][c] (LayerNorm'd)
// UPS=true: input is x (64x64), value = bilinear-upsampled on the fly.
// Block: 256 threads, 32 pixels; 8 thread-groups x 32 channels each.
// ---------------------------------------------------------------------------
template <bool UPS>
__global__ void lnT_kernel(const float* __restrict__ src,
                           const float* __restrict__ lw, const float* __restrict__ lb,
                           bf16_t* __restrict__ rawT, bf16_t* __restrict__ lnT) {
    const int t = threadIdx.x;
    const int p = t & 31, cg = t >> 5;           // pixel-in-tile, channel group
    const int b = blockIdx.z;
    const int n0 = blockIdx.x * 32;
    const int n = n0 + p;
    const int h2 = n >> 7, w2 = n & 127;

    __shared__ bf16_t raw_s[32 * 264];           // padded rows (264) vs bank conflicts
    __shared__ bf16_t ln_s [32 * 264];
    __shared__ float sum_s[8][32], sq_s[8][32];
    __shared__ float mu_s[32], rs_s[32];

    float sum = 0.0f, sq = 0.0f;
    for (int j = 0; j < 32; ++j) {
        int c = cg * 32 + j;
        float v;
        if (UPS) v = bilin_up(src, b, c, h2, w2);
        else     v = src[(((size_t)b * NC + c) << LOG2PIX) + n];
        sum += v; sq += v * v;
        raw_s[p * 264 + c] = (bf16_t)v;
    }
    sum_s[cg][p] = sum; sq_s[cg][p] = sq;
    __syncthreads();
    if (t < 32) {
        float s = 0.0f, q = 0.0f;
        for (int g = 0; g < 8; ++g) { s += sum_s[g][t]; q += sq_s[g][t]; }
        float mu  = s * (1.0f / 256.0f);
        float var = q * (1.0f / 256.0f) - mu * mu;
        mu_s[t] = mu;
        rs_s[t] = rsqrtf(var + 1e-6f);
    }
    __syncthreads();
    float mu = mu_s[p], rs = rs_s[p];
    for (int j = 0; j < 32; ++j) {
        int c = cg * 32 + j;
        float v = (float)raw_s[p * 264 + c];
        ln_s[p * 264 + c] = (bf16_t)(((v - mu) * rs) * lw[c] + lb[c]);
    }
    __syncthreads();
    // coalesced copy-out: row = pixel, 8 threads x 64B per 512B row
    {
        int row = t >> 3, seg = t & 7;
        const uint4* rsrc = (const uint4*)(raw_s + row * 264 + seg * 32);
        const uint4* lsrc = (const uint4*)(ln_s  + row * 264 + seg * 32);
        size_t gbase = ((size_t)(b << LOG2PIX) + n0 + row) * NC + seg * 32;
        uint4* rdst = (uint4*)(rawT + gbase);
        uint4* ldst = (uint4*)(lnT  + gbase);
        #pragma unroll
        for (int q2 = 0; q2 < 4; ++q2) { rdst[q2] = rsrc[q2]; ldst[q2] = lsrc[q2]; }
    }
}

// ---------------------------------------------------------------------------
// Q GEMM: out[b][o][n] (bf16, channel-major) = W(256x256) x X^T + bias
// Block: 128 thr = 4 waves, block tile 64x64, wave tile 32x32, K=256
// ---------------------------------------------------------------------------
__global__ void gemm_qT(const bf16_t* __restrict__ A, const bf16_t* __restrict__ BT,
                        const float* __restrict__ bias, bf16_t* __restrict__ out) {
    const int lane = threadIdx.x & 31, wid = threadIdx.x >> 5;
    const int b = blockIdx.z;
    const int m0 = blockIdx.y * 64 + (wid >> 1) * 32;
    const int n0 = blockIdx.x * 64 + (wid & 1) * 32;
    const bf16_t* Bb = BT + ((size_t)b << LOG2PIX) * NC;
    v8f acc[2][2] = {};
    for (int k0 = 0; k0 < 256; k0 += 32) {
        v16bf a0 = load_a_frag(A, 256, m0,      k0, lane);
        v16bf a1 = load_a_frag(A, 256, m0 + 16, k0, lane);
        v16bf b0 = load_b_frag(Bb, 256, n0,      k0, lane);
        v16bf b1 = load_b_frag(Bb, 256, n0 + 16, k0, lane);
        acc[0][0] = wmma_bf16(a0, b0, acc[0][0]);
        acc[0][1] = wmma_bf16(a0, b1, acc[0][1]);
        acc[1][0] = wmma_bf16(a1, b0, acc[1][0]);
        acc[1][1] = wmma_bf16(a1, b1, acc[1][1]);
    }
    bf16_t* ob = out + (size_t)b * NC * NPIX;
    const int nl = lane & 15, mo = (lane < 16) ? 0 : 8;
    #pragma unroll
    for (int mt = 0; mt < 2; ++mt)
        #pragma unroll
        for (int nt = 0; nt < 2; ++nt) {
            int n  = n0 + nt * 16 + nl;
            int mb = m0 + mt * 16 + mo;
            #pragma unroll
            for (int r = 0; r < 8; ++r)
                ob[(size_t)(mb + r) * NPIX + n] = (bf16_t)(acc[mt][nt][r] + bias[mb + r]);
        }
}

// ---------------------------------------------------------------------------
// V GEMM: same compute, stores TRANSPOSED into Vstack[b][head][n][64] bf16
// (voff=0 -> Vg slots d=0..31, voff=32 -> Vd slots d=32..63)
// ---------------------------------------------------------------------------
__global__ void gemm_vstack(const bf16_t* __restrict__ A, const bf16_t* __restrict__ BT,
                            const float* __restrict__ bias, bf16_t* __restrict__ Vst,
                            int voff) {
    const int lane = threadIdx.x & 31, wid = threadIdx.x >> 5;
    const int b = blockIdx.z;
    const int m0 = blockIdx.y * 64 + (wid >> 1) * 32;
    const int n0 = blockIdx.x * 64 + (wid & 1) * 32;
    const bf16_t* Bb = BT + ((size_t)b << LOG2PIX) * NC;
    v8f acc[2][2] = {};
    for (int k0 = 0; k0 < 256; k0 += 32) {
        v16bf a0 = load_a_frag(A, 256, m0,      k0, lane);
        v16bf a1 = load_a_frag(A, 256, m0 + 16, k0, lane);
        v16bf b0 = load_b_frag(Bb, 256, n0,      k0, lane);
        v16bf b1 = load_b_frag(Bb, 256, n0 + 16, k0, lane);
        acc[0][0] = wmma_bf16(a0, b0, acc[0][0]);
        acc[0][1] = wmma_bf16(a0, b1, acc[0][1]);
        acc[1][0] = wmma_bf16(a1, b0, acc[1][0]);
        acc[1][1] = wmma_bf16(a1, b1, acc[1][1]);
    }
    const int nl = lane & 15, mo = (lane < 16) ? 0 : 8;
    #pragma unroll
    for (int mt = 0; mt < 2; ++mt)
        #pragma unroll
        for (int nt = 0; nt < 2; ++nt) {
            int n  = n0 + nt * 16 + nl;
            int ob = m0 + mt * 16 + mo;          // output channel base, 8 contiguous
            int head = ob >> 5;
            int d0 = (ob & 31) + voff;
            v8bf pk;
            #pragma unroll
            for (int r = 0; r < 8; ++r) pk[r] = (bf16_t)(acc[mt][nt][r] + bias[ob + r]);
            *(v8bf*)(Vst + (((size_t)(b * NHEAD + head) << LOG2PIX) + n) * 64 + d0) = pk;
        }
}

// ---------------------------------------------------------------------------
// Per-row sum of squares over n (for folded L2 normalization). grid.x: 0..2047
// ---------------------------------------------------------------------------
__global__ void rowsq_kernel(const bf16_t* __restrict__ Qd, const bf16_t* __restrict__ Qg,
                             float* __restrict__ od, float* __restrict__ og) {
    int row = blockIdx.x;
    const bf16_t* src; float* dst;
    if (row < NB * NC) { src = Qd; dst = od; }
    else               { src = Qg; dst = og; row -= NB * NC; }
    const bf16_t* p = src + ((size_t)row << LOG2PIX);
    const int t = threadIdx.x;
    float s = 0.0f;
    for (int i = 0; i < 8; ++i) {
        v8bf v = *(const v8bf*)(p + ((size_t)(i * 256 + t)) * 8);
        #pragma unroll
        for (int j = 0; j < 8; ++j) { float f = (float)v[j]; s += f * f; }
    }
    __shared__ float red[256];
    red[t] = s; __syncthreads();
    for (int off = 128; off > 0; off >>= 1) {
        if (t < off) red[t] += red[t + off];
        __syncthreads();
    }
    if (t == 0) dst[row] = red[0];
}

// ---------------------------------------------------------------------------
// Gram: gram[bh][c][d] += sum_n Qd[c,n]*Qg[d,n]  (split-K WMMA + f32 atomics)
// grid (32 bh, 32 splitK), block 128 = 4 waves (2x2 tiles of 16x16)
// ---------------------------------------------------------------------------
#define GRAM_SPLITK 32
__global__ void gram_kernel(const bf16_t* __restrict__ Qd, const bf16_t* __restrict__ Qg,
                            float* __restrict__ gram) {
    const int lane = threadIdx.x & 31, wid = threadIdx.x >> 5;
    const int bh = blockIdx.x;
    const int kstart = blockIdx.y * (NPIX / GRAM_SPLITK);
    const bf16_t* Ab = Qd + ((size_t)(bh * HC) << LOG2PIX);
    const bf16_t* Bb = Qg + ((size_t)(bh * HC) << LOG2PIX);
    const int m0 = (wid >> 1) * 16, n0 = (wid & 1) * 16;
    v8f acc = {};
    for (int i = 0; i < (NPIX / GRAM_SPLITK) / 32; ++i) {
        int k0 = kstart + i * 32;
        acc = wmma_bf16(load_a_frag(Ab, NPIX, m0, k0, lane),
                        load_b_frag(Bb, NPIX, n0, k0, lane), acc);
    }
    float* g = gram + (size_t)bh * (HC * HC);
    const int nl = lane & 15, mo = (lane < 16) ? 0 : 8;
    #pragma unroll
    for (int r = 0; r < 8; ++r)
        atomicAdd(&g[(m0 + mo + r) * HC + n0 + nl], acc[r]);
}

// ---------------------------------------------------------------------------
// Softmax over d with folded L2 norms + build Acomb[bh][c][0..63] = [beta*A | gamma*A]
// grid 32 (bh), block 32 (one wave, thread = c row)
// ---------------------------------------------------------------------------
__global__ void softmax_acomb(const float* __restrict__ gram,
                              const float* __restrict__ rq_d, const float* __restrict__ rq_g,
                              const float* __restrict__ scale,
                              const float* __restrict__ beta, const float* __restrict__ gamma,
                              bf16_t* __restrict__ Acomb) {
    const int bh = blockIdx.x;
    const int h = bh & 7;
    const int c = threadIdx.x;
    __shared__ float ng_s[HC];
    ng_s[c] = fmaxf(sqrtf(rq_g[bh * HC + c]), 1e-12f);
    __syncthreads();
    float nd = fmaxf(sqrtf(rq_d[bh * HC + c]), 1e-12f);
    float sc = scale[h];
    const float* g = gram + (size_t)bh * (HC * HC) + c * HC;
    float v[HC];
    float mx = -3.0e38f;
    for (int d = 0; d < HC; ++d) {
        v[d] = g[d] * sc / (nd * ng_s[d]);
        mx = fmaxf(mx, v[d]);
    }
    float s = 0.0f;
    for (int d = 0; d < HC; ++d) { v[d] = __expf(v[d] - mx); s += v[d]; }
    float inv = 1.0f / s;
    int cg = h * HC + c;
    float bb = beta[cg], gg = gamma[cg];
    bf16_t* dst = Acomb + ((size_t)bh * HC + c) * 64;
    for (int d = 0; d < HC; ++d) {
        float a = v[d] * inv;
        dst[d]      = (bf16_t)(bb * a);
        dst[32 + d] = (bf16_t)(gg * a);
    }
}

// ---------------------------------------------------------------------------
// Mixing GEMM + residual: y[b][c][n] = upsample(x) + res + Acomb(32x64) x Vstack^T
// grid (128 n-tiles, 32 bh), block 128 = 4 waves, each wave 32x32 tile, K=64
// ---------------------------------------------------------------------------
__global__ void mix_kernel(const bf16_t* __restrict__ Acomb, const bf16_t* __restrict__ Vst,
                           const float* __restrict__ x, const float* __restrict__ res,
                           float* __restrict__ y) {
    const int lane = threadIdx.x & 31, wid = threadIdx.x >> 5;
    const int bh = blockIdx.y, b = bh >> 3, h = bh & 7;
    const int n0 = blockIdx.x * 128 + wid * 32;
    const bf16_t* Ab = Acomb + (size_t)bh * HC * 64;
    const bf16_t* Bb = Vst + ((size_t)bh << LOG2PIX) * 64;
    v8f acc[2][2] = {};
    #pragma unroll
    for (int k0 = 0; k0 < 64; k0 += 32) {
        v16bf a0 = load_a_frag(Ab, 64, 0,  k0, lane);
        v16bf a1 = load_a_frag(Ab, 64, 16, k0, lane);
        v16bf b0 = load_b_frag(Bb, 64, n0,      k0, lane);
        v16bf b1 = load_b_frag(Bb, 64, n0 + 16, k0, lane);
        acc[0][0] = wmma_bf16(a0, b0, acc[0][0]);
        acc[0][1] = wmma_bf16(a0, b1, acc[0][1]);
        acc[1][0] = wmma_bf16(a1, b0, acc[1][0]);
        acc[1][1] = wmma_bf16(a1, b1, acc[1][1]);
    }
    const int nl = lane & 15, mo = (lane < 16) ? 0 : 8;
    #pragma unroll
    for (int mt = 0; mt < 2; ++mt)
        #pragma unroll
        for (int nt = 0; nt < 2; ++nt) {
            int n = n0 + nt * 16 + nl;
            int hh = n >> 7, ww = n & 127;
            #pragma unroll
            for (int r = 0; r < 8; ++r) {
                int cl = mt * 16 + mo + r;
                int c  = h * HC + cl;
                size_t idx = (((size_t)b * NC + c) << LOG2PIX) + n;
                y[idx] = acc[mt][nt][r] + bilin_up(x, b, c, hh, ww) + res[idx];
            }
        }
}

// ---------------------------------------------------------------------------
// Depthwise 3x3 (SAME, zero pad) + BatchNorm(eval); write TRANSPOSED zT[b][n][c] bf16
// block 256 = 32 (w) x 8 (c-thread, 4 channels each); LDS transpose 32w x 32c
// ---------------------------------------------------------------------------
__global__ void dw_kernel(const float* __restrict__ y, const float* __restrict__ dww,
                          const float* __restrict__ bng, const float* __restrict__ bnb,
                          const float* __restrict__ bnm, const float* __restrict__ bnv,
                          bf16_t* __restrict__ zT) {
    const int t = threadIdx.x, tx = t & 31, ty = t >> 5;
    const int wt = blockIdx.x & 3, h = blockIdx.x >> 2;
    const int ct = blockIdx.y, b = blockIdx.z;
    const int w = wt * 32 + tx;
    __shared__ bf16_t zt[32 * 36];
    #pragma unroll
    for (int j = 0; j < 4; ++j) {
        int cl = ty * 4 + j;
        int c  = ct * 32 + cl;
        const float* yp = y + (((size_t)b * NC + c) << LOG2PIX);
        const float* wp = dww + c * 9;
        float accv = 0.0f;
        #pragma unroll
        for (int kh = 0; kh < 3; ++kh) {
            int hh = h + kh - 1;
            if ((unsigned)hh < 128u) {
                #pragma unroll
                for (int kw = 0; kw < 3; ++kw) {
                    int ww = w + kw - 1;
                    if ((unsigned)ww < 128u)
                        accv += yp[hh * 128 + ww] * wp[kh * 3 + kw];
                }
            }
        }
        float rs  = rsqrtf(bnv[c] + 1e-5f);
        float val = (accv - bnm[c]) * rs * bng[c] + bnb[c];
        zt[tx * 36 + cl] = (bf16_t)val;
    }
    __syncthreads();
    int row = t >> 3, seg = t & 7;                      // row = w-in-tile, seg*4 channels
    uint2 v2 = *(const uint2*)(zt + row * 36 + seg * 4);
    size_t n = (size_t)h * 128 + wt * 32 + row;
    *(uint2*)(zT + (((size_t)b << LOG2PIX) + n) * NC + ct * 32 + seg * 4) = v2;
}

// ---------------------------------------------------------------------------
// Pointwise GEMM + ReLU6: out[b][o][n] = clip(pw(256x256) x zT, 0, 6)  (f32 out)
// ---------------------------------------------------------------------------
__global__ void gemm_pw(const bf16_t* __restrict__ A, const bf16_t* __restrict__ BT,
                        float* __restrict__ out) {
    const int lane = threadIdx.x & 31, wid = threadIdx.x >> 5;
    const int b = blockIdx.z;
    const int m0 = blockIdx.y * 64 + (wid >> 1) * 32;
    const int n0 = blockIdx.x * 64 + (wid & 1) * 32;
    const bf16_t* Bb = BT + ((size_t)b << LOG2PIX) * NC;
    v8f acc[2][2] = {};
    for (int k0 = 0; k0 < 256; k0 += 32) {
        v16bf a0 = load_a_frag(A, 256, m0,      k0, lane);
        v16bf a1 = load_a_frag(A, 256, m0 + 16, k0, lane);
        v16bf b0 = load_b_frag(Bb, 256, n0,      k0, lane);
        v16bf b1 = load_b_frag(Bb, 256, n0 + 16, k0, lane);
        acc[0][0] = wmma_bf16(a0, b0, acc[0][0]);
        acc[0][1] = wmma_bf16(a0, b1, acc[0][1]);
        acc[1][0] = wmma_bf16(a1, b0, acc[1][0]);
        acc[1][1] = wmma_bf16(a1, b1, acc[1][1]);
    }
    float* ob = out + (size_t)b * NC * NPIX;
    const int nl = lane & 15, mo = (lane < 16) ? 0 : 8;
    #pragma unroll
    for (int mt = 0; mt < 2; ++mt)
        #pragma unroll
        for (int nt = 0; nt < 2; ++nt) {
            int n  = n0 + nt * 16 + nl;
            int mb = m0 + mt * 16 + mo;
            #pragma unroll
            for (int r = 0; r < 8; ++r)
                ob[(size_t)(mb + r) * NPIX + n] = fminf(fmaxf(acc[mt][nt][r], 0.0f), 6.0f);
        }
}

// ---------------------------------------------------------------------------
// Launch
// ---------------------------------------------------------------------------
extern "C" void kernel_launch(void* const* d_in, const int* in_sizes, int n_in,
                              void* d_out, int out_size, void* d_ws, size_t ws_size,
                              hipStream_t stream) {
    const float* x      = (const float*)d_in[0];
    const float* res    = (const float*)d_in[1];
    const float* ln_d_w = (const float*)d_in[2];
    const float* ln_d_b = (const float*)d_in[3];
    const float* ln_g_w = (const float*)d_in[4];
    const float* ln_g_b = (const float*)d_in[5];
    const float* w_dp1  = (const float*)d_in[6];
    const float* b_dp1  = (const float*)d_in[7];
    const float* w_gp1  = (const float*)d_in[8];
    const float* b_gp1  = (const float*)d_in[9];
    const float* w_dp2  = (const float*)d_in[10];
    const float* b_dp2  = (const float*)d_in[11];
    const float* w_gp2  = (const float*)d_in[12];
    const float* b_gp2  = (const float*)d_in[13];
    const float* scale  = (const float*)d_in[14];
    const float* beta   = (const float*)d_in[15];
    const float* gamma  = (const float*)d_in[16];
    const float* dw_w   = (const float*)d_in[17];
    const float* bn_g   = (const float*)d_in[18];
    const float* bn_b   = (const float*)d_in[19];
    const float* bn_m   = (const float*)d_in[20];
    const float* bn_v   = (const float*)d_in[21];
    const float* pw_w   = (const float*)d_in[22];

    char* ws = (char*)d_ws;
    const size_t MB = 1024ull * 1024ull;
    bf16_t* xdlnT = (bf16_t*)(ws + 0 * MB);     // 32 MB  [b][n][c] LN(x_up)
    bf16_t* xupT  = (bf16_t*)(ws + 32 * MB);    // 32 MB  [b][n][c] x_up
    bf16_t* xglnT = (bf16_t*)(ws + 64 * MB);    // 32 MB  [b][n][c] LN(res)
    bf16_t* resT  = (bf16_t*)(ws + 96 * MB);    // 32 MB  [b][n][c] res
    bf16_t* Qd    = (bf16_t*)(ws + 128 * MB);   // 32 MB  [b][c][n]
    bf16_t* Qg    = (bf16_t*)(ws + 160 * MB);   // 32 MB  [b][c][n]
    bf16_t* Vst   = (bf16_t*)(ws + 192 * MB);   // 64 MB  [b][head][n][64] = [Vg|Vd]
    bf16_t* wbf   = (bf16_t*)(ws + 256 * MB);   // 640 KB weights bf16 x5
    float*  gram  = (float*) (ws + 257 * MB);   // 128 KB
    float*  rq_d  = (float*) (ws + 258 * MB);   // 4 KB
    float*  rq_g  = (float*) (ws + 258 * MB + 4096);
    bf16_t* Acomb = (bf16_t*)(ws + 259 * MB);   // 128 KB
    float*  ybuf  = (float*) (ws + 128 * MB);   // 64 MB, aliases Qd/Qg (dead by then)
    bf16_t* zT    = (bf16_t*)(ws + 0 * MB);     // 32 MB, aliases xdlnT (dead by then)

    // 1) weights -> bf16, zero gram (re-done each call -> deterministic)
    prep_kernel<<<1408, 256, 0, stream>>>(w_dp1, w_gp1, w_dp2, w_gp2, pw_w, wbf, gram);
    // 2) LN + transpose (upsample fused for the x path)
    lnT_kernel<true ><<<dim3(NPIX / 32, 1, NB), 256, 0, stream>>>(x,   ln_d_w, ln_d_b, xupT, xdlnT);
    lnT_kernel<false><<<dim3(NPIX / 32, 1, NB), 256, 0, stream>>>(res, ln_g_w, ln_g_b, resT, xglnT);
    // 3) Q projections (channel-major out, for gram fragments)
    gemm_qT<<<dim3(256, 4, NB), 128, 0, stream>>>(wbf + 0 * 65536, xdlnT, b_dp1, Qd);
    gemm_qT<<<dim3(256, 4, NB), 128, 0, stream>>>(wbf + 1 * 65536, xglnT, b_gp1, Qg);
    // 4) V projections, stored transposed into stacked V (Vg=0..31, Vd=32..63)
    gemm_vstack<<<dim3(256, 4, NB), 128, 0, stream>>>(wbf + 3 * 65536, resT, b_gp2, Vst, 0);
    gemm_vstack<<<dim3(256, 4, NB), 128, 0, stream>>>(wbf + 2 * 65536, xupT, b_dp2, Vst, 32);
    // 5) L2-norm row sums of squares (folded into softmax)
    rowsq_kernel<<<2 * NB * NC, 256, 0, stream>>>(Qd, Qg, rq_d, rq_g);
    // 6) Gram matrix via split-K WMMA + atomics
    gram_kernel<<<dim3(NB * NHEAD, GRAM_SPLITK), 128, 0, stream>>>(Qd, Qg, gram);
    // 7) softmax + combined [beta*A | gamma*A]
    softmax_acomb<<<NB * NHEAD, 32, 0, stream>>>(gram, rq_d, rq_g, scale, beta, gamma, Acomb);
    // 8) attention apply + residual (x_up recomputed inline) -> y
    mix_kernel<<<dim3(NPIX / 128, NB * NHEAD), 128, 0, stream>>>(Acomb, Vst, x, res, ybuf);
    // 9) depthwise 3x3 + BN -> transposed bf16
    dw_kernel<<<dim3(4 * 128, 8, NB), 256, 0, stream>>>(ybuf, dw_w, bn_g, bn_b, bn_m, bn_v, zT);
    // 10) pointwise conv + ReLU6 -> d_out (f32, NCHW)
    gemm_pw<<<dim3(256, 4, NB), 128, 0, stream>>>(wbf + 4 * 65536, zT, (float*)d_out);

    (void)in_sizes; (void)n_in; (void)out_size; (void)ws_size;
}
